// SSIM_18073222382098
// MI455X (gfx1250) — matrix operand order repaired and checked by reading, not compiled, not run on hardware
//
#include <hip/hip_runtime.h>

typedef __attribute__((ext_vector_type(16))) _Float16 v16h;
typedef __attribute__((ext_vector_type(8)))  float    v8f;
typedef __attribute__((ext_vector_type(2)))  _Float16 h2;

#define IMG      512
#define OUTD     502
#define NPLANES  96                 // 32 batch * 3 channels
#define WAVES_PB 2
#define STRIPS_PER_PLANE 128        // 16 row-strips * 8 col-strips (strip = 32 x 64 outputs)
#define TOTAL_WAVES (NPLANES * STRIPS_PER_PLANE)   // 12288
#define NBLOCKS     (TOTAL_WAVES / WAVES_PB)       // 6144

// per-wave LDS layout (bytes)
#define XROW_STRIDE 176             // 160B data (80 halves) + pad; 16 rows -> 16 distinct bank groups
#define XTILE_BYTES (48 * XROW_STRIDE)   // 8448  (rows 0..47, 42 live)
#define YOFF        XTILE_BYTES
#define TMPOFF      (2 * XTILE_BYTES)    // 16896
#define TROW_STRIDE 112                  // 96B data (48 halves per column) + pad
#define TMP_BYTES   (16 * TROW_STRIDE)   // 1792, double buffered
#define WAVE_LDS    (TMPOFF + 2 * TMP_BYTES)   // 20480

struct alignas(16) F4 { float x, y, z, w; };
struct alignas(8)  U2 { unsigned x, y; };
struct alignas(16) U4 { unsigned x, y, z, w; };
struct alignas(32) U4x2 { U4 lo, hi; };

__device__ __forceinline__ unsigned pack2(float a, float b) {
  h2 h; h.x = (_Float16)a; h.y = (_Float16)b;
  return __builtin_bit_cast(unsigned, h);
}
__device__ __forceinline__ v16h make_v16h(U4 lo, U4 hi) {
  U4x2 p{lo, hi};
  return __builtin_bit_cast(v16h, p);
}

__global__ __launch_bounds__(WAVES_PB * 32)
void ssim_strip_kernel(const float* __restrict__ X, const float* __restrict__ Y,
                       const float* __restrict__ WIN, float* __restrict__ partial)
{
  __shared__ alignas(16) char lds[WAVES_PB * WAVE_LDS];
  __shared__ alignas(16) _Float16 tBH[512];   // horizontal banded B, per-lane operand layout
  __shared__ alignas(16) _Float16 tGA[512];   // vertical   banded A, per-lane operand layout
  __shared__ float g_lds[11];
  __shared__ float redbuf[WAVES_PB];

  const int tid   = threadIdx.x;
  const int wave  = tid >> 5;
  const int lane  = tid & 31;
  const int lhalf = lane >> 4;
  const int lmod  = lane & 15;
  char* wl = lds + wave * WAVE_LDS;

  // 1D gaussian from window diagonal: g[k] = sqrt(k2d[k][k])
  if (tid < 11) g_lds[tid] = sqrtf(WIN[tid * 12]);

  // ---- strip coordinates: 32 output rows x 64 output cols per wave ----
  int wgl   = blockIdx.x * WAVES_PB + wave;
  int plane = wgl >> 7;                 // / STRIPS_PER_PLANE
  int rem   = wgl & 127;
  int ti    = rem >> 3, sj = rem & 7;
  int gy    = ti * 32, gx = sj * 64;    // strip origin (input == output for VALID conv)
  const float* xp = X + (size_t)plane * (IMG * IMG);
  const float* yp = Y + (size_t)plane * (IMG * IMG);

  // ---- phase 1: stage x and y strips as f16 (48 rows x 80 halves; rows >=42 zero) ----
  #pragma unroll
  for (int half48 = 0; half48 < 2; ++half48) {
    int r = half48 * 32 + lane;               // rows 0..31, then 32..47 (lanes 0..15)
    if (r < 48) {
      int grow = gy + r;
      bool rowok = (r < 42) && (grow < IMG);
      const float* xr = xp + (size_t)grow * IMG;
      const float* yr = yp + (size_t)grow * IMG;
      char* rb = wl + r * XROW_STRIDE;
      if (rowok && (gx + 80) <= IMG) {
        // fast path: whole 80-col row in bounds (7/8 of col-strips)
        #pragma unroll 4
        for (int gq = 0; gq < 20; ++gq) {
          F4 xv = *(const F4*)(xr + gx + gq * 4);
          F4 yv = *(const F4*)(yr + gx + gq * 4);
          U2 px{pack2(xv.x, xv.y), pack2(xv.z, xv.w)};
          U2 py{pack2(yv.x, yv.y), pack2(yv.z, yv.w)};
          *(U2*)(rb + gq * 8)        = px;
          *(U2*)(rb + YOFF + gq * 8) = py;
        }
      } else {
        // edge path: per-group bounds (groups of 4 are all-in or all-out vs IMG)
        for (int gq = 0; gq < 20; ++gq) {
          int c0 = gx + gq * 4;
          F4 xv{0.f, 0.f, 0.f, 0.f}, yv{0.f, 0.f, 0.f, 0.f};
          if (rowok && (c0 + 3) < IMG) {
            xv = *(const F4*)(xr + c0);
            yv = *(const F4*)(yr + c0);
          }
          U2 px{pack2(xv.x, xv.y), pack2(xv.z, xv.w)};
          U2 py{pack2(yv.x, yv.y), pack2(yv.z, yv.w)};
          *(U2*)(rb + gq * 8)        = px;
          *(U2*)(rb + YOFF + gq * 8) = py;
        }
      }
    }
  }

  // ---- build banded operand tables once per block (rolled loop) ----
  __syncthreads();                         // g_lds visible
  for (int e = tid; e < 1024; e += WAVES_PB * 32) {
    int isGA = e >> 9;
    int idx  = e & 511;
    int l = idx >> 4, t = idx & 15;
    int col = l & 15, hf = l >> 4;
    int k = (t < 8) ? (hf * 8 + t) : (16 + hf * 8 + (t - 8));
    int d  = k - col;
    int dc = d < 0 ? 0 : (d > 10 ? 10 : d);
    _Float16 v = (d >= 0 && d <= 10) ? (_Float16)g_lds[dc] : (_Float16)0.0f;
    if (isGA) tGA[idx] = v; else tBH[idx] = v;
  }
  __syncthreads();                         // tables visible

  const char* pBH = (const char*)tBH + lane * 32;
  const char* pGA = (const char*)tGA + lane * 32;
  v16h BH = make_v16h(*(const U4*)pBH, *(const U4*)(pBH + 16));
  v16h GA = make_v16h(*(const U4*)pGA, *(const U4*)(pGA + 16));

  // ---- phase 2: 4 col-tiles x (2 row-groups) per strip, separable conv via WMMA ----
  const float c1 = 1.0e-4f, c2 = 9.0e-4f;
  float psum = 0.0f;

  for (int tj = 0; tj < 4; ++tj) {
    // A-operands for x and y rows 0..47; squares/cross derived via v_pk_mul_f16
    v16h Ax[3], Ay[3];
    #pragma unroll
    for (int h = 0; h < 3; ++h) {
      int row = h * 16 + lmod;
      const char* bx = wl + row * XROW_STRIDE + tj * 32 + lhalf * 16;
      Ax[h] = make_v16h(*(const U4*)bx, *(const U4*)(bx + 32));
      const char* by = bx + YOFF;
      Ay[h] = make_v16h(*(const U4*)by, *(const U4*)(by + 32));
    }

    v8f acc[5][2];
    #pragma unroll
    for (int q = 0; q < 5; ++q) {
      char* tb = wl + TMPOFF + (q & 1) * TMP_BYTES;   // double-buffered tmp
      #pragma unroll
      for (int h = 0; h < 3; ++h) {
        v16h A;
        if      (q == 0) A = Ax[h];
        else if (q == 1) A = Ay[h];
        else if (q == 2) A = Ax[h] * Ax[h];           // v_pk_mul_f16
        else if (q == 3) A = Ay[h] * Ay[h];
        else             A = Ax[h] * Ay[h];
        v8f d = {};
        d = __builtin_amdgcn_wmma_f32_16x16x32_f16(
                false, A, false, BH, (short)0, d, false, false);
        // transpose-store tmp as f16: lane owns column lmod, rows h*16 + 8*lhalf + v
        U4 st{pack2(d[0], d[1]), pack2(d[2], d[3]),
              pack2(d[4], d[5]), pack2(d[6], d[7])};
        *(U4*)(tb + lmod * TROW_STRIDE + h * 32 + lhalf * 16) = st;
      }
      __builtin_amdgcn_wave_barrier();     // LDS in-order per wave; pin scheduler (RAW)
      #pragma unroll
      for (int oi = 0; oi < 2; ++oi) {     // vertical: B = tmp rows 16*oi .. 16*oi+31
        const char* bp = tb + lmod * TROW_STRIDE + oi * 32 + lhalf * 16;
        v16h B = make_v16h(*(const U4*)bp, *(const U4*)(bp + 32));
        v8f c = {};
        acc[q][oi] = __builtin_amdgcn_wmma_f32_16x16x32_f16(
                         false, GA, false, B, (short)0, c, false, false);
      }
      __builtin_amdgcn_wave_barrier();     // WAR vs reuse of this buffer (q+2)
    }

    // elementwise SSIM + valid-region mask (all accs share the C/D layout).
    // Denominators >= c1 > 0; 1-ULP v_rcp_f32 is well inside the f16-conv error budget.
    #pragma unroll
    for (int oi = 0; oi < 2; ++oi) {
      #pragma unroll
      for (int v = 0; v < 8; ++v) {
        int i = oi * 16 + lhalf * 8 + v;
        int j = tj * 16 + lmod;
        bool ok = (gy + i < OUTD) && (gx + j < OUTD);
        float mx  = acc[0][oi][v], my  = acc[1][oi][v];
        float exx = acc[2][oi][v], eyy = acc[3][oi][v], exy = acc[4][oi][v];
        float mx2 = mx * mx, my2 = my * my, mxy = mx * my;
        float sxx = exx - mx2, syy = eyy - my2, sxy = exy - mxy;
        float cs  = (2.0f * sxy + c2) * __builtin_amdgcn_rcpf(sxx + syy + c2);
        float ss  = (2.0f * mxy + c1) * __builtin_amdgcn_rcpf(mx2 + my2 + c1) * cs;
        if (ok) psum += ss;
      }
    }
  }

  // ---- deterministic reduction: wave -> block -> d_ws ----
  #pragma unroll
  for (int off = 16; off > 0; off >>= 1)
    psum += __shfl_xor(psum, off, 32);
  if (lane == 0) redbuf[wave] = psum;
  __syncthreads();
  if (tid == 0)
    partial[blockIdx.x] = redbuf[0] + redbuf[1];
}

__global__ __launch_bounds__(256)
void ssim_reduce_kernel(const float* __restrict__ partial, float* __restrict__ out)
{
  __shared__ float buf[256];
  float s = 0.0f;
  for (int i = threadIdx.x; i < NBLOCKS; i += 256) s += partial[i];
  buf[threadIdx.x] = s;
  __syncthreads();
  #pragma unroll
  for (int off = 128; off > 0; off >>= 1) {
    if (threadIdx.x < off) buf[threadIdx.x] += buf[threadIdx.x + off];
    __syncthreads();
  }
  if (threadIdx.x == 0)
    out[0] = buf[0] * (1.0f / 24192384.0f);   // 32*3*502*502
}

extern "C" void kernel_launch(void* const* d_in, const int* in_sizes, int n_in,
                              void* d_out, int out_size, void* d_ws, size_t ws_size,
                              hipStream_t stream) {
  const float* x   = (const float*)d_in[0];
  const float* y   = (const float*)d_in[1];
  const float* win = (const float*)d_in[2];
  float* partial = (float*)d_ws;             // NBLOCKS floats = 24 KB scratch
  ssim_strip_kernel<<<NBLOCKS, WAVES_PB * 32, 0, stream>>>(x, y, win, partial);
  ssim_reduce_kernel<<<1, 256, 0, stream>>>(partial, (float*)d_out);
}
